// FrameExtraction_42520176230826
// MI455X (gfx1250) — matrix operand order repaired and checked by reading, not compiled, not run on hardware
//
#include <hip/hip_runtime.h>
#include <hip/hip_bf16.h>
#include <math.h>

// ---------------------------------------------------------------------------
// CDNA5 WMMA types / bf16 helpers
// ---------------------------------------------------------------------------
typedef __attribute__((ext_vector_type(16))) __bf16 v16bf;
typedef __attribute__((ext_vector_type(8)))  __bf16 v8bf;
typedef __attribute__((ext_vector_type(8)))  float  v8f;

__device__ __forceinline__ __bf16 f2bf(float f) {
  union { float f; unsigned u; } c; c.f = f;
  unsigned r = c.u + 0x7FFFu + ((c.u >> 16) & 1u);   // round-to-nearest-even
  unsigned short h = (unsigned short)(r >> 16);
  __bf16 b; __builtin_memcpy(&b, &h, 2);
  return b;
}
__device__ __forceinline__ float bf2f(__bf16 b) {
  unsigned short h; __builtin_memcpy(&h, &b, 2);
  union { unsigned u; float f; } c; c.u = ((unsigned)h) << 16;
  return c.f;
}

// ---------------------------------------------------------------------------
// Implicit-GEMM convolution.
//   X  : NHWC bf16, channel stride Cin (multiple of 32, zero-padded)
//   Wt : [co][tap][ci] bf16, ci stride Cin
//   Y  : NHWC fp32, channel stride outCtot, written at channel offset outOff
// One wave = 32(Cout) x 32(pixel) tile, 4 x v_wmma_f32_16x16x32_bf16 per
// 32-deep K step. All fragment feeds are 16B vector loads:
//   A: lanes<16 hold K 0..7 & 16..23, lanes>=16 hold K 8..15 & 24..31
//   B: lanes<16 hold K 0..7 & 16..23, lanes>=16 hold K 8..15 & 24..31
// ---------------------------------------------------------------------------
__global__ void conv_wmma_kernel(const __bf16* __restrict__ X,
                                 const __bf16* __restrict__ Wt,
                                 const float* __restrict__ bias,
                                 float* __restrict__ Y,
                                 int Nimg, int Cin, int H, int W,
                                 int Cout, int outCtot, int outOff,
                                 int KK, int stride, int pad) {
  const int OH = (H + 2 * pad - KK) / stride + 1;
  const int OW = (W + 2 * pad - KK) / stride + 1;
  const int P  = OH * OW;                       // multiple of 32 at all call sites
  const int tilesM = Cout >> 5;
  const int tilesP = P >> 5;
  const int wavesPerImg = tilesM * tilesP;
  const int totalWaves  = Nimg * wavesPerImg;

  const int wid = blockIdx.x * (blockDim.x >> 5) + (threadIdx.x >> 5);
  if (wid >= totalWaves) return;

  const int lane = threadIdx.x & 31;
  const int hi   = lane >> 4;
  const int ln   = lane & 15;

  const int img   = wid / wavesPerImg;
  const int rem   = wid - img * wavesPerImg;
  const int mBase = (rem / tilesP) * 32;
  const int pBase = (rem % tilesP) * 32;

  const int KK2 = KK * KK;
  const __bf16* __restrict__ xb = X + (size_t)img * H * W * Cin;
  const __bf16 BZ = f2bf(0.f);
  v8bf zz;
#pragma unroll
  for (int e = 0; e < 8; ++e) zz[e] = BZ;

  const int p0 = pBase + ln;
  const int p1 = p0 + 16;
  const int oh0 = p0 / OW, ow0 = p0 - oh0 * OW;
  const int oh1 = p1 / OW, ow1 = p1 - oh1 * OW;

  v8f acc00 = {}; v8f acc01 = {}; v8f acc10 = {}; v8f acc11 = {};

  for (int kh = 0; kh < KK; ++kh) {
    for (int kw = 0; kw < KK; ++kw) {
      const int tap = kh * KK + kw;
      const int ih0 = oh0 * stride + kh - pad, iw0 = ow0 * stride + kw - pad;
      const int ih1 = oh1 * stride + kh - pad, iw1 = ow1 * stride + kw - pad;
      const bool v0 = ((unsigned)ih0 < (unsigned)H) && ((unsigned)iw0 < (unsigned)W);
      const bool v1 = ((unsigned)ih1 < (unsigned)H) && ((unsigned)iw1 < (unsigned)W);
      const __bf16* __restrict__ xp0 = xb + ((size_t)ih0 * W + iw0) * Cin + hi * 8;
      const __bf16* __restrict__ xp1 = xb + ((size_t)ih1 * W + iw1) * Cin + hi * 8;
      const __bf16* __restrict__ wa0 = Wt + ((size_t)(mBase + ln)      * KK2 + tap) * Cin + hi * 8;
      const __bf16* __restrict__ wa1 = Wt + ((size_t)(mBase + 16 + ln) * KK2 + tap) * Cin + hi * 8;

      for (int ci0 = 0; ci0 < Cin; ci0 += 32) {
        if (v0 && ci0 + 32 < Cin) {
          __builtin_prefetch(xp0 + ci0 + 32, 0, 1);   // global_prefetch_b8
        }
        v8bf a0l = *(const v8bf*)(wa0 + ci0);
        v8bf a0h = *(const v8bf*)(wa0 + ci0 + 16);
        v8bf a1l = *(const v8bf*)(wa1 + ci0);
        v8bf a1h = *(const v8bf*)(wa1 + ci0 + 16);
        v8bf b0l = zz, b0h = zz, b1l = zz, b1h = zz;
        if (v0) {
          b0l = *(const v8bf*)(xp0 + ci0);
          b0h = *(const v8bf*)(xp0 + ci0 + 16);
        }
        if (v1) {
          b1l = *(const v8bf*)(xp1 + ci0);
          b1h = *(const v8bf*)(xp1 + ci0 + 16);
        }
        v16bf a0 = __builtin_shufflevector(a0l, a0h, 0,1,2,3,4,5,6,7,8,9,10,11,12,13,14,15);
        v16bf a1 = __builtin_shufflevector(a1l, a1h, 0,1,2,3,4,5,6,7,8,9,10,11,12,13,14,15);
        v16bf b0 = __builtin_shufflevector(b0l, b0h, 0,1,2,3,4,5,6,7,8,9,10,11,12,13,14,15);
        v16bf b1 = __builtin_shufflevector(b1l, b1h, 0,1,2,3,4,5,6,7,8,9,10,11,12,13,14,15);
        acc00 = __builtin_amdgcn_wmma_f32_16x16x32_bf16(false, a0, false, b0, (short)0, acc00, false, false);
        acc01 = __builtin_amdgcn_wmma_f32_16x16x32_bf16(false, a0, false, b1, (short)0, acc01, false, false);
        acc10 = __builtin_amdgcn_wmma_f32_16x16x32_bf16(false, a1, false, b0, (short)0, acc10, false, false);
        acc11 = __builtin_amdgcn_wmma_f32_16x16x32_bf16(false, a1, false, b1, (short)0, acc11, false, false);
      }
    }
  }

  // D: lanes<16 -> M=r, lanes>=16 -> M=r+8 ; 8 rows = 8 consecutive co in NHWC
  if (bias) {
    const v8f bz0 = *(const v8f*)(bias + mBase + hi * 8);
    const v8f bz1 = *(const v8f*)(bias + mBase + 16 + hi * 8);
    acc00 += bz0; acc01 += bz0; acc10 += bz1; acc11 += bz1;
  }
  float* __restrict__ y0 = Y + ((size_t)img * P + p0) * outCtot + outOff + mBase + hi * 8;
  float* __restrict__ y1 = Y + ((size_t)img * P + p1) * outCtot + outOff + mBase + hi * 8;
  *(v8f*)(y0)      = acc00;
  *(v8f*)(y0 + 16) = acc10;
  *(v8f*)(y1)      = acc01;
  *(v8f*)(y1 + 16) = acc11;
}

// ---------------------------------------------------------------------------
// BatchNorm statistics over NHWC fp32: one block per channel, LDS tree reduce.
// ---------------------------------------------------------------------------
__global__ void bn_stats_kernel(const float* __restrict__ X,
                                float* __restrict__ mean, float* __restrict__ rstd,
                                int N, int C, int HW) {
  const int c = blockIdx.x;
  float s = 0.f, ss = 0.f;
  const int tot = N * HW;
  for (int i = threadIdx.x; i < tot; i += blockDim.x) {
    const float v = X[(size_t)i * C + c];
    s += v; ss += v * v;
  }
  __shared__ float sh1[256];
  __shared__ float sh2[256];
  sh1[threadIdx.x] = s; sh2[threadIdx.x] = ss;
  __syncthreads();
  for (int off = 128; off > 0; off >>= 1) {
    if ((int)threadIdx.x < off) {
      sh1[threadIdx.x] += sh1[threadIdx.x + off];
      sh2[threadIdx.x] += sh2[threadIdx.x + off];
    }
    __syncthreads();
  }
  if (threadIdx.x == 0) {
    const float inv = 1.f / (float)tot;
    const float m = sh1[0] * inv;
    const float var = fmaxf(sh2[0] * inv - m * m, 0.f);
    mean[c] = m;
    rstd[c] = rsqrtf(var + 1e-5f);
  }
}

__device__ __forceinline__ void stv(float* y, size_t i, float v)  { y[i] = v; }
__device__ __forceinline__ void stv(__bf16* y, size_t i, float v) { y[i] = f2bf(v); }

// X NHWC fp32 (C chans); res NHWC bf16 same shape; out NHWC slice (nchwOut=0)
// or NCHW fp32 (nchwOut=1, outCtot==C, outOff==0).
template <typename TOut>
__global__ void bn_apply_kernel(const float* __restrict__ X,
                                const float* __restrict__ mean,
                                const float* __restrict__ rstd,
                                const __bf16* __restrict__ res,
                                TOut* __restrict__ Y,
                                int N, int C, int HW, int outCtot, int outOff,
                                int relu, int nchwOut) {
  const size_t total = (size_t)N * C * HW;
  size_t i = (size_t)blockIdx.x * blockDim.x + threadIdx.x;
  if (i >= total) return;
  const int c = (int)(i % C);
  const size_t q = i / C;              // n*HW + p
  float v = (X[i] - mean[c]) * rstd[c];
  if (res) v += bf2f(res[i]);
  if (relu) v = fmaxf(v, 0.f);
  if (nchwOut) {
    const size_t n = q / HW, p = q - n * HW;
    stv(Y, ((size_t)n * C + c) * HW + p, v);
  } else {
    stv(Y, q * outCtot + outOff + c, v);
  }
}

// OIHW fp32 -> [co][tap][ciP] bf16 (ci zero-padded to CinP)
__global__ void cvt_wt_kernel(const float* __restrict__ src, __bf16* __restrict__ dst,
                              int Cout, int Cin, int CinP, int KK2) {
  const size_t total = (size_t)Cout * KK2 * CinP;
  size_t i = (size_t)blockIdx.x * blockDim.x + threadIdx.x;
  if (i >= total) return;
  const int ci  = (int)(i % CinP);
  const int tap = (int)((i / CinP) % KK2);
  const int co  = (int)(i / ((size_t)CinP * KK2));
  dst[i] = (ci < Cin) ? f2bf(src[((size_t)co * Cin + ci) * KK2 + tap]) : f2bf(0.f);
}

// NCHW fp32 -> NHWC bf16 with channel pad to CP
__global__ void nchw2nhwc_bf_kernel(const float* __restrict__ src, __bf16* __restrict__ dst,
                                    int N, int C, int CP, int HW) {
  const size_t total = (size_t)N * HW * CP;
  size_t i = (size_t)blockIdx.x * blockDim.x + threadIdx.x;
  if (i >= total) return;
  const int c = (int)(i % CP);
  const size_t q = i / CP;
  const size_t n = q / HW, p = q - n * HW;
  dst[i] = (c < C) ? f2bf(src[((size_t)n * C + c) * HW + p]) : f2bf(0.f);
}

// NHWC fp32 -> NCHW fp32
__global__ void nhwc2nchw_f32_kernel(const float* __restrict__ src, float* __restrict__ dst,
                                     int N, int C, int HW) {
  const size_t total = (size_t)N * C * HW;
  size_t i = (size_t)blockIdx.x * blockDim.x + threadIdx.x;
  if (i >= total) return;
  const int p = (int)(i % HW);
  const int c = (int)((i / HW) % C);
  const int n = (int)(i / ((size_t)C * HW));
  dst[i] = src[((size_t)n * HW + p) * C + c];
}

// ---------------------------------------------------------------------------
// Haar DWT level: NCHW fp32 (N,C,H,W) -> (N,4C,H/2,W/2)
// ---------------------------------------------------------------------------
__global__ void haar_kernel(const float* __restrict__ X, float* __restrict__ Y,
                            int N, int C, int H, int W) {
  const int OH = H >> 1, OW = W >> 1;
  const size_t total = (size_t)N * C * OH * OW;
  size_t i = (size_t)blockIdx.x * blockDim.x + threadIdx.x;
  if (i >= total) return;
  const int ow = (int)(i % OW);
  const int oh = (int)((i / OW) % OH);
  const int c  = (int)((i / ((size_t)OW * OH)) % C);
  const int n  = (int)(i / ((size_t)C * OH * OW));
  const float* xp = X + (((size_t)(n * C + c) * H + 2 * oh) * W + 2 * ow);
  const float a = xp[0], b = xp[1], cc = xp[W], d = xp[W + 1];
  float* yp = Y + (((size_t)(n * C * 4 + c * 4)) * OH + oh) * OW + ow;
  const size_t bs = (size_t)OH * OW;
  yp[0]      = (a + b + cc + d) * 0.5f;
  yp[bs]     = (a + b - cc - d) * 0.5f;
  yp[2 * bs] = (a - b + cc - d) * 0.5f;
  yp[3 * bs] = (a - b - cc + d) * 0.5f;
}

// ---------------------------------------------------------------------------
// ConvTranspose2d 4x4/s2/p1. X: NCHW fp32, W:(Cin,Cout,4,4), Y: NHWC fp32.
// ---------------------------------------------------------------------------
__global__ void convt_kernel(const float* __restrict__ X,
                             const float* __restrict__ Wt,
                             const float* __restrict__ bias,
                             float* __restrict__ Y,
                             int N, int Cin, int H, int W, int Cout) {
  const int OH = 2 * H, OW = 2 * W;
  const size_t total = (size_t)N * Cout * OH * OW;
  size_t i = (size_t)blockIdx.x * blockDim.x + threadIdx.x;
  if (i >= total) return;
  const int co = (int)(i % Cout);
  const size_t q = i / Cout;                 // n*OH*OW + oh*OW + ow
  const int ow = (int)(q % OW);
  const int oh = (int)((q / OW) % OH);
  const int n  = (int)(q / ((size_t)OH * OW));
  float acc = bias[co];
  for (int kh = 0; kh < 4; ++kh) {
    const int qh = oh + kh - 2;
    if (qh < 0 || (qh & 1) || (qh >> 1) >= H) continue;
    const int ih = qh >> 1;
    for (int kw = 0; kw < 4; ++kw) {
      const int qw = ow + kw - 2;
      if (qw < 0 || (qw & 1) || (qw >> 1) >= W) continue;
      const int iw = qw >> 1;
      const float* xp = X + (((size_t)(n * Cin)) * H + ih) * W + iw;
      const float* wp = Wt + (size_t)co * 16 + ((3 - kh) * 4 + (3 - kw));
      for (int ci = 0; ci < Cin; ++ci)
        acc += wp[(size_t)ci * Cout * 16] * xp[(size_t)ci * H * W];
    }
  }
  Y[q * Cout + co] = acc;
}

// Spatial mean over HW (NHWC fp32); one block per (n,c).
__global__ void smean_kernel(const float* __restrict__ X, float* __restrict__ out,
                             int C, int HW) {
  const int n = blockIdx.x / C;
  const int c = blockIdx.x - n * C;
  const float* xp = X + (size_t)n * HW * C + c;
  float s = 0.f;
  for (int i = threadIdx.x; i < HW; i += blockDim.x) s += xp[(size_t)i * C];
  __shared__ float sh[256];
  sh[threadIdx.x] = s;
  __syncthreads();
  for (int off = 128; off > 0; off >>= 1) {
    if ((int)threadIdx.x < off) sh[threadIdx.x] += sh[threadIdx.x + off];
    __syncthreads();
  }
  if (threadIdx.x == 0) out[blockIdx.x] = sh[0] / (float)HW;
}

// Channel-attention MLP: s = sigmoid(W2 * relu(W1 * s0)) per image.
__global__ void ca_mlp_kernel(const float* __restrict__ s0,
                              const float* __restrict__ w1,
                              const float* __restrict__ w2,
                              float* __restrict__ s, int C, int Cm) {
  __shared__ float t[32];
  const int n = blockIdx.x;
  for (int j = threadIdx.x; j < Cm; j += blockDim.x) {
    float a = 0.f;
    for (int c = 0; c < C; ++c) a += w1[(size_t)j * C + c] * s0[(size_t)n * C + c];
    t[j] = fmaxf(a, 0.f);
  }
  __syncthreads();
  for (int c = threadIdx.x; c < C; c += blockDim.x) {
    float a = 0.f;
    for (int j = 0; j < Cm; ++j) a += w2[(size_t)c * Cm + j] * t[j];
    s[(size_t)n * C + c] = 1.f / (1.f + expf(-a));
  }
}

// y[i] = x[i] * s[n,c] ; NHWC fp32
__global__ void scale_mul_kernel(const float* __restrict__ X, const float* __restrict__ s,
                                 float* __restrict__ Y, int N, int C, int HW) {
  const size_t total = (size_t)N * C * HW;
  size_t i = (size_t)blockIdx.x * blockDim.x + threadIdx.x;
  if (i >= total) return;
  const int c = (int)(i % C);
  const int n = (int)(i / ((size_t)C * HW));
  Y[i] = X[i] * s[(size_t)n * C + c];
}

// Copy channel slice between NHWC bf16 tensors.
__global__ void copy_slice_bf_kernel(const __bf16* __restrict__ src, __bf16* __restrict__ dst,
                                     int N, int Csl, int HW,
                                     int srcCtot, int srcOff, int dstCtot, int dstOff) {
  const size_t total = (size_t)N * HW * Csl;
  size_t i = (size_t)blockIdx.x * blockDim.x + threadIdx.x;
  if (i >= total) return;
  const int c = (int)(i % Csl);
  const size_t q = i / Csl;
  dst[q * dstCtot + dstOff + c] = src[q * srcCtot + srcOff + c];
}

// ---------------------------------------------------------------------------
// Host orchestration
// ---------------------------------------------------------------------------
extern "C" void kernel_launch(void* const* d_in, const int* in_sizes, int n_in,
                              void* d_out, int out_size, void* d_ws, size_t ws_size,
                              hipStream_t stream) {
  (void)in_sizes; (void)n_in; (void)out_size; (void)ws_size;
  auto IN = [&](int i) { return (const float*)d_in[i]; };

  const float* X      = IN(0);
  const float* Wconv1 = IN(1);
  const float* Wct1   = IN(2);  const float* Bct1  = IN(3);
  const float* Wct2a  = IN(4);  const float* Bct2a = IN(5);
  const float* Wct2b  = IN(6);  const float* Bct2b = IN(7);
  const float* Wfwx   = IN(8);  const float* Bfwx  = IN(9);
  const float* Wfwx2  = IN(10); const float* Bfwx2 = IN(11);
  const float* L1w1[4] = {IN(12), IN(14), IN(16), IN(18)};
  const float* L1w3[4] = {IN(13), IN(15), IN(17), IN(19)};
  const float* L2w1[4] = {IN(20), IN(23), IN(25), IN(27)};
  const float* L2w3[4] = {IN(21), IN(24), IN(26), IN(28)};
  const float* L2wds   = IN(22);
  const float* L3w1[4] = {IN(29), IN(32), IN(34), IN(36)};
  const float* L3w3[4] = {IN(30), IN(33), IN(35), IN(37)};
  const float* L3wds   = IN(31);
  const float* Wfuse1  = IN(38);
  const float* Wfuse2  = IN(39);
  const float* Wfuse3  = IN(40);
  const float* Wca1    = IN(41);
  const float* Wca2    = IN(42);
  const float* Wproj   = IN(43);

  // ---- workspace ----
  float* ws = (float*)d_ws;
  size_t off = 0;
  auto allocF = [&](size_t n) { float* p = ws + off; off += n; return p; };
  const size_t BIGE = (size_t)8 * 128 * 128 * 128; // 16,777,216 elements
  float* RAW1 = allocF(BIGE);                      // conv raw outputs, NHWC fp32
  float* RAW2 = allocF(BIGE);
  float* CAT3 = allocF((size_t)8 * 64 * 64 * 448); // [f1|f2|f3] NHWC fp32
  float* B2A  = allocF((size_t)8 * 32 * 64 * 64);  // NCHW fp32 (feeds convt)
  float* XL1  = allocF((size_t)8 * 12 * 64 * 64);  // NCHW fp32
  float* XL2  = allocF((size_t)8 * 48 * 32 * 32);  // NCHW fp32
  float* MEAN = allocF(512);
  float* RSTD = allocF(512);
  float* S0   = allocF(8 * 448);
  float* SS   = allocF(8 * 448);
  __bf16* bw = (__bf16*)(ws + off);
  size_t boff = 0;
  auto allocB = [&](size_t n) { __bf16* p = bw + boff; boff += (n + 7) & ~(size_t)7; return p; };
  __bf16* CAT1B = allocB(BIGE);                    // NHWC 128ch: [h|xw] then [x1|xw]
  __bf16* TB    = allocB(BIGE);
  __bf16* XA    = allocB(BIGE);
  __bf16* XB    = allocB(BIGE);
  __bf16* IDNB  = allocB(BIGE);
  __bf16* XINB  = allocB((size_t)8 * 128 * 128 * 32); // input, 3->32 padded NHWC

  auto ew = [&](size_t total) { return (unsigned)((total + 255) / 256); };

  // bf16 weight mirrors, [co][tap][ciP]
  auto cvtW = [&](const float* src, int Cout, int Cin, int KK2) {
    const int CinP = (Cin + 31) & ~31;
    const size_t n = (size_t)Cout * KK2 * CinP;
    __bf16* p = allocB(n);
    cvt_wt_kernel<<<ew(n), 256, 0, stream>>>(src, p, Cout, Cin, CinP, KK2);
    return p;
  };
  __bf16* bWconv1 = cvtW(Wconv1, 64, 3, 9);
  __bf16* bWfwx   = cvtW(Wfwx,   64, 128, 9);
  __bf16* bWfwx2  = cvtW(Wfwx2,  64, 128, 9);
  __bf16* bL1w1[4]; __bf16* bL1w3[4];
  for (int b = 0; b < 4; ++b) { bL1w1[b] = cvtW(L1w1[b], 64, 64, 1); bL1w3[b] = cvtW(L1w3[b], 64, 64, 1); }
  __bf16* bL2w1[4]; __bf16* bL2w3[4];
  bL2w1[0] = cvtW(L2w1[0], 128, 64, 1);  bL2w3[0] = cvtW(L2w3[0], 128, 128, 1);
  for (int b = 1; b < 4; ++b) { bL2w1[b] = cvtW(L2w1[b], 128, 128, 1); bL2w3[b] = cvtW(L2w3[b], 128, 128, 1); }
  __bf16* bL2wds = cvtW(L2wds, 128, 64, 1);
  __bf16* bL3w1[4]; __bf16* bL3w3[4];
  bL3w1[0] = cvtW(L3w1[0], 256, 128, 1); bL3w3[0] = cvtW(L3w3[0], 256, 256, 1);
  for (int b = 1; b < 4; ++b) { bL3w1[b] = cvtW(L3w1[b], 256, 256, 1); bL3w3[b] = cvtW(L3w3[b], 256, 256, 1); }
  __bf16* bL3wds  = cvtW(L3wds, 256, 128, 1);
  __bf16* bWfuse1 = cvtW(Wfuse1, 64, 64, 9);
  __bf16* bWfuse2 = cvtW(Wfuse2, 128, 128, 9);
  __bf16* bWfuse3 = cvtW(Wfuse3, 256, 256, 9);
  __bf16* bWproj  = cvtW(Wproj, 256, 448, 9);

  auto launchConv = [&](const __bf16* Xi, const __bf16* Wt, const float* bias, float* Y,
                        int N, int Cin, int H, int W, int Cout, int outCtot, int outOff,
                        int KK, int stride, int pad) {
    const int OH = (H + 2 * pad - KK) / stride + 1;
    const int OW = (W + 2 * pad - KK) / stride + 1;
    const int waves = N * (Cout >> 5) * ((OH * OW) >> 5);
    conv_wmma_kernel<<<(waves + 7) / 8, 256, 0, stream>>>(
        Xi, Wt, bias, Y, N, Cin, H, W, Cout, outCtot, outOff, KK, stride, pad);
  };
  auto bnToBF = [&](const float* Xr, int N, int C, int HW, const __bf16* res,
                    __bf16* Y, int outCtot, int outOff, int relu) {
    bn_stats_kernel<<<C, 256, 0, stream>>>(Xr, MEAN, RSTD, N, C, HW);
    bn_apply_kernel<__bf16><<<ew((size_t)N * C * HW), 256, 0, stream>>>(
        Xr, MEAN, RSTD, res, Y, N, C, HW, outCtot, outOff, relu, 0);
  };
  auto bnToNCHW = [&](const float* Xr, int N, int C, int HW, float* Y, int relu) {
    bn_stats_kernel<<<C, 256, 0, stream>>>(Xr, MEAN, RSTD, N, C, HW);
    bn_apply_kernel<float><<<ew((size_t)N * C * HW), 256, 0, stream>>>(
        Xr, MEAN, RSTD, nullptr, Y, N, C, HW, C, 0, relu, 1);
  };

  // ---- Stage A: wavelet branch -> xw = CAT1B[:, :, 64:128] ----
  haar_kernel<<<ew((size_t)8 * 3 * 64 * 64), 256, 0, stream>>>(X, XL1, 8, 3, 128, 128);
  haar_kernel<<<ew((size_t)8 * 12 * 32 * 32), 256, 0, stream>>>(XL1, XL2, 8, 12, 64, 64);

  convt_kernel<<<ew((size_t)8 * 32 * 128 * 128), 256, 0, stream>>>(XL1, Wct1, Bct1, RAW1, 8, 12, 64, 64, 32);
  bnToBF(RAW1, 8, 32, 16384, nullptr, CAT1B, 128, 64, 1);          // b1

  convt_kernel<<<ew((size_t)8 * 32 * 64 * 64), 256, 0, stream>>>(XL2, Wct2a, Bct2a, RAW2, 8, 48, 32, 32, 32);
  bnToNCHW(RAW2, 8, 32, 4096, B2A, 1);
  convt_kernel<<<ew((size_t)8 * 32 * 128 * 128), 256, 0, stream>>>(B2A, Wct2b, Bct2b, RAW1, 8, 32, 64, 64, 32);
  bnToBF(RAW1, 8, 32, 16384, nullptr, CAT1B, 128, 96, 1);          // b2

  // ---- Stage B: stem ----
  nchw2nhwc_bf_kernel<<<ew((size_t)8 * 16384 * 32), 256, 0, stream>>>(X, XINB, 8, 3, 32, 16384);
  launchConv(XINB, bWconv1, nullptr, RAW1, 8, 32, 128, 128, 64, 64, 0, 3, 1, 1);
  bnToBF(RAW1, 8, 64, 16384, nullptr, CAT1B, 128, 0, 1);           // h
  launchConv(CAT1B, bWfwx, Bfwx, RAW1, 8, 128, 128, 128, 64, 64, 0, 3, 1, 1);
  // h0 fp32 NHWC -> bf16 NHWC (straight copy via slice-less bn? use copy path)
  bn_stats_kernel<<<1, 256, 0, stream>>>(RAW1, MEAN, RSTD, 0, 1, 0); // no-op warmup (C=1,N*HW=0)
  {
    // plain fp32->bf16 convert, same layout
    const size_t n = (size_t)8 * 16384 * 64;
    // reuse cvt_wt_kernel as flat converter: Cout=n, Cin=1, CinP=1, KK2=1
    cvt_wt_kernel<<<ew(n), 256, 0, stream>>>(RAW1, XA, (int)(n), 1, 1, 1);
  }

  // ---- layer1 (4 blocks, 64ch @128^2) ----
  __bf16* xcur = XA;
  for (int b = 0; b < 4; ++b) {
    __bf16* dst = (xcur == XA) ? XB : XA;
    launchConv(xcur, bL1w1[b], nullptr, RAW1, 8, 64, 128, 128, 64, 64, 0, 1, 1, 0);
    bnToBF(RAW1, 8, 64, 16384, nullptr, TB, 64, 0, 1);
    launchConv(TB, bL1w3[b], nullptr, RAW2, 8, 64, 128, 128, 64, 64, 0, 1, 1, 0);
    bnToBF(RAW2, 8, 64, 16384, xcur, dst, 64, 0, 1);
    xcur = dst;
  }
  copy_slice_bf_kernel<<<ew((size_t)8 * 16384 * 64), 256, 0, stream>>>(
      xcur, CAT1B, 8, 64, 16384, 64, 0, 128, 0);
  launchConv(CAT1B, bWfwx2, Bfwx2, RAW1, 8, 128, 128, 128, 64, 64, 0, 3, 1, 1);
  {
    const size_t n = (size_t)8 * 16384 * 64;
    cvt_wt_kernel<<<ew(n), 256, 0, stream>>>(RAW1, XB, (int)(n), 1, 1, 1);
  }

  // f1 -> CAT3[:, :, 0:64]
  launchConv(XB, bWfuse1, nullptr, CAT3, 8, 64, 128, 128, 64, 448, 0, 3, 2, 1);

  // ---- layer2 (64 -> 128 @128^2) ----
  xcur = XB;
  launchConv(xcur, bL2w1[0], nullptr, RAW1, 8, 64, 128, 128, 128, 128, 0, 1, 1, 0);
  bnToBF(RAW1, 8, 128, 16384, nullptr, TB, 128, 0, 1);
  launchConv(xcur, bL2wds, nullptr, RAW2, 8, 64, 128, 128, 128, 128, 0, 1, 1, 0);
  bnToBF(RAW2, 8, 128, 16384, nullptr, IDNB, 128, 0, 0);
  launchConv(TB, bL2w3[0], nullptr, RAW2, 8, 128, 128, 128, 128, 128, 0, 1, 1, 0);
  bnToBF(RAW2, 8, 128, 16384, IDNB, XA, 128, 0, 1);
  xcur = XA;
  for (int b = 1; b < 4; ++b) {
    __bf16* dst = (xcur == XA) ? XB : XA;
    launchConv(xcur, bL2w1[b], nullptr, RAW1, 8, 128, 128, 128, 128, 128, 0, 1, 1, 0);
    bnToBF(RAW1, 8, 128, 16384, nullptr, TB, 128, 0, 1);
    launchConv(TB, bL2w3[b], nullptr, RAW2, 8, 128, 128, 128, 128, 128, 0, 1, 1, 0);
    bnToBF(RAW2, 8, 128, 16384, xcur, dst, 128, 0, 1);
    xcur = dst;
  }
  launchConv(xcur, bWfuse2, nullptr, CAT3, 8, 128, 128, 128, 128, 448, 64, 3, 2, 1);

  // ---- layer3 (128 -> 256, stride2 -> 64^2) ----
  {
    __bf16* dst = (xcur == XA) ? XB : XA;
    launchConv(xcur, bL3w1[0], nullptr, RAW1, 8, 128, 128, 128, 256, 256, 0, 1, 2, 0);
    bnToBF(RAW1, 8, 256, 4096, nullptr, TB, 256, 0, 1);
    launchConv(xcur, bL3wds, nullptr, RAW2, 8, 128, 128, 128, 256, 256, 0, 1, 2, 0);
    bnToBF(RAW2, 8, 256, 4096, nullptr, IDNB, 256, 0, 0);
    launchConv(TB, bL3w3[0], nullptr, RAW2, 8, 256, 64, 64, 256, 256, 0, 1, 1, 0);
    bnToBF(RAW2, 8, 256, 4096, IDNB, dst, 256, 0, 1);
    xcur = dst;
  }
  for (int b = 1; b < 4; ++b) {
    __bf16* dst = (xcur == XA) ? XB : XA;
    launchConv(xcur, bL3w1[b], nullptr, RAW1, 8, 256, 64, 64, 256, 256, 0, 1, 1, 0);
    bnToBF(RAW1, 8, 256, 4096, nullptr, TB, 256, 0, 1);
    launchConv(TB, bL3w3[b], nullptr, RAW2, 8, 256, 64, 64, 256, 256, 0, 1, 1, 0);
    bnToBF(RAW2, 8, 256, 4096, xcur, dst, 256, 0, 1);
    xcur = dst;
  }
  launchConv(xcur, bWfuse3, nullptr, CAT3, 8, 256, 64, 64, 256, 448, 192, 3, 1, 1);

  // ---- channel attention + bn + projection ----
  smean_kernel<<<8 * 448, 256, 0, stream>>>(CAT3, S0, 448, 4096);
  ca_mlp_kernel<<<8, 256, 0, stream>>>(S0, Wca1, Wca2, SS, 448, 28);
  scale_mul_kernel<<<ew((size_t)8 * 448 * 4096), 256, 0, stream>>>(CAT3, SS, RAW1, 8, 448, 4096);
  bnToBF(RAW1, 8, 448, 4096, nullptr, TB, 448, 0, 0);
  launchConv(TB, bWproj, nullptr, RAW2, 8, 448, 64, 64, 256, 256, 0, 3, 1, 1);
  nhwc2nchw_f32_kernel<<<ew((size_t)8 * 256 * 4096), 256, 0, stream>>>(
      RAW2, (float*)d_out, 8, 256, 4096);
}
